// Attention_70660801954538
// MI455X (gfx1250) — matrix operand order repaired and checked by reading, not compiled, not run on hardware
//
#include <hip/hip_runtime.h>
#include <hip/hip_bf16.h>

typedef __attribute__((ext_vector_type(16))) _Float16 v16h;
typedef __attribute__((ext_vector_type(8)))  _Float16 h8;
typedef __attribute__((ext_vector_type(8)))  float    v8f;
typedef __attribute__((ext_vector_type(4)))  float    f4;

#define BD 8
#define NN 1024
#define CC 768
#define HH 12
#define HDIM 64
#define SCALE 0.125f
#define VTS 40  // Vt LDS row stride in halves (x2B = 80B, multiple of 16B for b128 loads)

__device__ __forceinline__ v8f wmma16(v16h a, v16h b, v8f c) {
  return __builtin_amdgcn_wmma_f32_16x16x32_f16(false, a, false, b, (short)0, c, false, false);
}

// A fragment 16x32 f16: lane<16 -> m=lane, k={0..7,16..23}; lane>=16 -> m=lane-16, k={8..15,24..31}
__device__ __forceinline__ v16h ldfrag_a16(const _Float16* tile, int ld, int lane) {
  int m = lane & 15;
  int ks = (lane & 16) ? 8 : 0;
  const _Float16* p = tile + (size_t)m * ld + ks;
  union { v16h v; h8 q[2]; } u;
  u.q[0] = *(const h8*)(p);
  u.q[1] = *(const h8*)(p + 16);
  return u.v;
}

// Same fragment but sourced from f32 with in-register conversion
__device__ __forceinline__ v16h ldfrag_a32(const float* tile, int ld, int lane) {
  int m = lane & 15;
  int ks = (lane & 16) ? 8 : 0;
  const float* p = tile + (size_t)m * ld + ks;
  f4 x0 = *(const f4*)(p);
  f4 x1 = *(const f4*)(p + 4);
  f4 x2 = *(const f4*)(p + 16);
  f4 x3 = *(const f4*)(p + 20);
  union { v16h v; _Float16 e[16]; } u;
#pragma unroll
  for (int i = 0; i < 4; ++i) {
    u.e[i]      = (_Float16)x0[i];
    u.e[4 + i]  = (_Float16)x1[i];
    u.e[8 + i]  = (_Float16)x2[i];
    u.e[12 + i] = (_Float16)x3[i];
  }
  return u.v;
}

// B fragment 32x16 f16, B stored [N,K] row-major (contraction contiguous):
// lane -> n=lane&15, k = lane<16 ? {0..7,16..23} : {8..15,24..31}
__device__ __forceinline__ v16h ldfrag_b16(const _Float16* tile, int ld, int lane) {
  int n = lane & 15;
  int ks = (lane & 16) ? 8 : 0;
  const _Float16* p = tile + (size_t)n * ld + ks;
  union { v16h v; h8 q[2]; } u;
  u.q[0] = *(const h8*)(p);
  u.q[1] = *(const h8*)(p + 16);
  return u.v;
}

__global__ void cvt_f32_f16(const float* __restrict__ in, _Float16* __restrict__ out, int n) {
  int i = blockIdx.x * blockDim.x + threadIdx.x;
  if (i < n) out[i] = (_Float16)in[i];
}

// out[m,n] = alpha * sum_k A[m,k]*B[n,k]  (+ bias[n]); both operands contraction-contiguous (NT)
// Block = 128 threads = 4 waves; block tile 64x64; each wave does 16x64 (4 accumulators).
template <bool OUT16, bool BIAS, bool A32>
__global__ __launch_bounds__(128) void gemm_nt(
    const void* __restrict__ Ap, const _Float16* __restrict__ Bm,
    void* __restrict__ Outp, const float* __restrict__ bias,
    int K, int lda, int ldb, int ldo,
    long long sA, long long sB, long long sO, float alpha) {
  const int lane = threadIdx.x & 31;
  const int wave = threadIdx.x >> 5;
  const int bm = blockIdx.y * 64 + wave * 16;
  const int bn = blockIdx.x * 64;
  const long long bz = blockIdx.z;

  v8f acc[4] = {v8f{}, v8f{}, v8f{}, v8f{}};
  const _Float16* Bb = Bm + (size_t)(bz * sB);
  if constexpr (A32) {
    const float* A = (const float*)Ap + (size_t)(bz * sA) + (size_t)bm * lda;
    for (int k0 = 0; k0 < K; k0 += 32) {
      v16h a = ldfrag_a32(A + k0, lda, lane);
#pragma unroll
      for (int j = 0; j < 4; ++j) {
        v16h bf = ldfrag_b16(Bb + (size_t)(bn + 16 * j) * ldb + k0, ldb, lane);
        acc[j] = wmma16(a, bf, acc[j]);
      }
    }
  } else {
    const _Float16* A = (const _Float16*)Ap + (size_t)(bz * sA) + (size_t)bm * lda;
    for (int k0 = 0; k0 < K; k0 += 32) {
      v16h a = ldfrag_a16(A + k0, lda, lane);
#pragma unroll
      for (int j = 0; j < 4; ++j) {
        v16h bf = ldfrag_b16(Bb + (size_t)(bn + 16 * j) * ldb + k0, ldb, lane);
        acc[j] = wmma16(a, bf, acc[j]);
      }
    }
  }
  const int n = lane & 15;
  const int mb = (lane & 16) ? 8 : 0;
#pragma unroll
  for (int j = 0; j < 4; ++j) {
    int col = bn + 16 * j + n;
    float bv = BIAS ? bias[col] : 0.0f;
#pragma unroll
    for (int r = 0; r < 8; ++r) {
      size_t idx = (size_t)(bz * sO) + (size_t)(bm + mb + r) * ldo + col;
      float val = acc[j][r] * alpha + bv;
      if constexpr (OUT16) ((_Float16*)Outp)[idx] = (_Float16)val;
      else                 ((float*)Outp)[idx] = val;
    }
  }
}

// aff_mask[b,nk] = sigmoid( (SCALE/H) * dot(extra_token[b,:], K[b,nk,:]) )
__global__ void aff_kernel(const float* __restrict__ et, const _Float16* __restrict__ K16,
                           float* __restrict__ aff) {
  int idx = blockIdx.x * blockDim.x + threadIdx.x;
  if (idx >= BD * NN) return;
  int b = idx >> 10, nk = idx & (NN - 1);
  const _Float16* kr = K16 + ((size_t)b * NN + nk) * CC;
  const float* e = et + (size_t)b * CC;
  float s = 0.0f;
  for (int c = 0; c < CC; ++c) s += e[c] * (float)kr[c];
  float x = s * (SCALE / (float)HH);
  aff[idx] = 1.0f / (1.0f + __expf(-x));
}

// One block per (16-query tile, head, batch). 128 threads = 4 waves.
// Phase 1: S = Q_h K_h^T * SCALE (WMMA, wave covers 256 key-cols)
// Phase 2: block softmax over 1024 keys, * aff_mask, -> P (f16, LDS)
// Phase 3: out = P @ V_h (WMMA, V tile transposed through LDS), wave covers 16 out-cols
__global__ __launch_bounds__(128) void attn_kernel(
    const _Float16* __restrict__ Q16, const _Float16* __restrict__ K16,
    const _Float16* __restrict__ V16, const float* __restrict__ aff_mask,
    _Float16* __restrict__ O16) {
  __shared__ alignas(16) float    S[16 * 1024];
  __shared__ alignas(16) _Float16 P[16 * 1024];
  __shared__ alignas(16) _Float16 Vt[64 * VTS];
  __shared__ float red[128];
  __shared__ float rowmax[16], rowsum[16];

  const int tid = threadIdx.x, lane = tid & 31, wave = tid >> 5;
  const int qt = blockIdx.x, h = blockIdx.y, b = blockIdx.z;

  const _Float16* Qh = Q16 + ((size_t)b * NN + (size_t)qt * 16) * CC + h * HDIM;
  const _Float16* Kh = K16 + (size_t)b * NN * CC + h * HDIM;
  const _Float16* Vh = V16 + (size_t)b * NN * CC + h * HDIM;

  v16h aq0 = ldfrag_a16(Qh, CC, lane);
  v16h aq1 = ldfrag_a16(Qh + 32, CC, lane);

  // ---- scores ----
  for (int nt = 0; nt < 16; ++nt) {
    int n0 = wave * 256 + nt * 16;
    v8f acc = {};
    v16h b0 = ldfrag_b16(Kh + (size_t)n0 * CC, CC, lane);
    v16h b1 = ldfrag_b16(Kh + (size_t)n0 * CC + 32, CC, lane);
    acc = wmma16(aq0, b0, acc);
    acc = wmma16(aq1, b1, acc);
    int n = lane & 15, mb = (lane & 16) ? 8 : 0;
#pragma unroll
    for (int r = 0; r < 8; ++r) S[(mb + r) * 1024 + n0 + n] = acc[r] * SCALE;
  }
  __syncthreads();

  // ---- softmax: 8 threads per row ----
  const int row = tid >> 3, sub = tid & 7;
  float m = -3.4e38f;
  for (int c = sub; c < 1024; c += 8) m = fmaxf(m, S[row * 1024 + c]);
  red[tid] = m;
  __syncthreads();
  if (tid < 16) {
    float mm = red[tid * 8];
    for (int i = 1; i < 8; ++i) mm = fmaxf(mm, red[tid * 8 + i]);
    rowmax[tid] = mm;
  }
  __syncthreads();
  float rm = rowmax[row];
  float s = 0.0f;
  for (int c = sub; c < 1024; c += 8) {
    float e = __expf(S[row * 1024 + c] - rm);
    S[row * 1024 + c] = e;
    s += e;
  }
  red[tid] = s;
  __syncthreads();
  if (tid < 16) {
    float ss = 0.0f;
    for (int i = 0; i < 8; ++i) ss += red[tid * 8 + i];
    rowsum[tid] = ss;
  }
  __syncthreads();
  {
    float inv = 1.0f / rowsum[row];
    const float* aff = aff_mask + (size_t)b * NN;
    for (int c = sub; c < 1024; c += 8)
      P[row * 1024 + c] = (_Float16)(aff[c] * S[row * 1024 + c] * inv);
  }
  __syncthreads();

  // ---- P @ V ----
  v8f acc = {};
  for (int kk = 0; kk < 1024; kk += 32) {
    // stage V[kk..kk+31, 0..63] transposed into Vt[d][k]
    int kr = tid >> 2, d0 = (tid & 3) * 16;
    const _Float16* src = Vh + (size_t)(kk + kr) * CC + d0;
    h8 x0 = *(const h8*)src;
    h8 x1 = *(const h8*)(src + 8);
#pragma unroll
    for (int i = 0; i < 8; ++i) {
      Vt[(d0 + i) * VTS + kr]     = x0[i];
      Vt[(d0 + 8 + i) * VTS + kr] = x1[i];
    }
    __syncthreads();
    v16h a  = ldfrag_a16(P + kk, 1024, lane);
    v16h bf = ldfrag_b16(Vt + (size_t)(wave * 16) * VTS, VTS, lane);
    acc = wmma16(a, bf, acc);
    __syncthreads();
  }
  {
    int n = lane & 15, mb = (lane & 16) ? 8 : 0;
    _Float16* Oh = O16 + ((size_t)b * NN + (size_t)qt * 16) * CC + h * HDIM + wave * 16;
#pragma unroll
    for (int r = 0; r < 8; ++r)
      Oh[(size_t)(mb + r) * CC + n] = (_Float16)acc[r];
  }
}

extern "C" void kernel_launch(void* const* d_in, const int* in_sizes, int n_in,
                              void* d_out, int out_size, void* d_ws, size_t ws_size,
                              hipStream_t stream) {
  const float* xq = (const float*)d_in[0];
  const float* xk = (const float*)d_in[1];
  const float* xv = (const float*)d_in[2];
  const float* et = (const float*)d_in[3];
  const float* Wq = (const float*)d_in[4];
  const float* Wk = (const float*)d_in[5];
  const float* Wv = (const float*)d_in[6];
  const float* Wp = (const float*)d_in[7];
  const float* bp = (const float*)d_in[8];
  float* out = (float*)d_out;

  const size_t WSZ = (size_t)CC * CC;            // 589824
  const size_t XSZ = (size_t)BD * NN * CC;       // 6291456
  _Float16* ws   = (_Float16*)d_ws;
  _Float16* wq16 = ws;
  _Float16* wk16 = wq16 + WSZ;
  _Float16* wv16 = wk16 + WSZ;
  _Float16* wp16 = wv16 + WSZ;
  _Float16* q16  = wp16 + WSZ;
  _Float16* k16  = q16 + XSZ;
  _Float16* v16  = k16 + XSZ;
  _Float16* o16  = v16 + XSZ;

  float* out_x    = out;                                   // [8,1024,768]
  float* out_attn = out + XSZ;                             // [8,1024,1024]
  float* out_aff  = out + XSZ + (size_t)BD * NN * NN / 8 * 8; // placed after attn
  out_aff = out + XSZ + (size_t)BD * NN * NN;              // [8,1024]

  // 1) weights -> f16
  {
    dim3 blk(256), g((unsigned)((WSZ + 255) / 256));
    cvt_f32_f16<<<g, blk, 0, stream>>>(Wq, wq16, (int)WSZ);
    cvt_f32_f16<<<g, blk, 0, stream>>>(Wk, wk16, (int)WSZ);
    cvt_f32_f16<<<g, blk, 0, stream>>>(Wv, wv16, (int)WSZ);
    cvt_f32_f16<<<g, blk, 0, stream>>>(Wp, wp16, (int)WSZ);
  }

  // 2) QKV projections: [8192x768] = x @ W^T, f32 A converted in-register, f16 out
  dim3 gp(CC / 64, (BD * NN) / 64, 1);  // (12,128)
  gemm_nt<true, false, true><<<gp, 128, 0, stream>>>(
      xq, wq16, q16, nullptr, CC, CC, CC, CC, 0, 0, 0, 1.0f);
  gemm_nt<true, false, true><<<gp, 128, 0, stream>>>(
      xk, wk16, k16, nullptr, CC, CC, CC, CC, 0, 0, 0, 1.0f);
  gemm_nt<true, false, true><<<gp, 128, 0, stream>>>(
      xv, wv16, v16, nullptr, CC, CC, CC, CC, 0, 0, 0, 1.0f);

  // 3) aff_mask -> d_out (read later by attn_kernel)
  aff_kernel<<<dim3((BD * NN) / 256), 256, 0, stream>>>(et, k16, out_aff);

  // 4) attn_save = (SCALE/H) * Q_full @ K_full^T  (per-head sum collapses to full dot)
  gemm_nt<false, false, false><<<dim3(NN / 64, NN / 64, BD), 128, 0, stream>>>(
      q16, k16, out_attn, nullptr, CC, CC, CC, NN,
      (long long)NN * CC, (long long)NN * CC, (long long)NN * NN, SCALE / (float)HH);

  // 5) attention per (q-tile, head, batch)
  attn_kernel<<<dim3(NN / 16, HH, BD), 128, 0, stream>>>(q16, k16, v16, out_aff, o16);

  // 6) x = attn_out @ Wp^T + bp
  gemm_nt<false, true, false><<<gp, 128, 0, stream>>>(
      o16, wp16, out_x, bp, CC, CC, CC, CC, 0, 0, 0, 1.0f);

  (void)in_sizes; (void)n_in; (void)out_size; (void)ws_size;
}
